// MultiHeadAttention_5506148073933
// MI455X (gfx1250) — matrix operand order repaired and checked by reading, not compiled, not run on hardware
//
#include <hip/hip_runtime.h>
#include <hip/hip_bf16.h>

// ---------------------------------------------------------------------------
// MHA forward for B=2, S=2048, D=768, H=12, DH=64 on gfx1250 (CDNA5).
// All matmuls on v_wmma_f32_16x16x32_bf16 (f32 accumulate).
// Round 4: ping-pong (unroll-by-2) GEMM K-loop -> no register-rotation movs.
// ---------------------------------------------------------------------------

constexpr int Bb = 2;
constexpr int Ss = 2048;
constexpr int Dd = 768;
constexpr int Hh = 12;
constexpr int DH = 64;
constexpr int Mm = Bb * Ss;          // 4096 token rows

typedef __attribute__((ext_vector_type(16))) __bf16 v16bf;
typedef __attribute__((ext_vector_type(8)))  __bf16 v8bf;
typedef __attribute__((ext_vector_type(8)))  float  v8f;
typedef __attribute__((ext_vector_type(4)))  int    v4i;

#if __has_builtin(__builtin_amdgcn_global_load_async_to_lds_b128)
#define HAVE_ASYNC_LDS 1
#else
#define HAVE_ASYNC_LDS 0
#endif

#if HAVE_ASYNC_LDS
#if __has_builtin(__builtin_amdgcn_s_wait_asynccnt)
#define ASYNC_WAIT(n) __builtin_amdgcn_s_wait_asynccnt(n)
#else
#define ASYNC_WAIT(n) asm volatile("s_wait_asynccnt %0" ::"i"(n) : "memory")
#endif
#endif

__device__ __forceinline__ __bf16 f2bf(float f) {
  unsigned u = __builtin_bit_cast(unsigned, f);
  unsigned r = u + 0x7FFFu + ((u >> 16) & 1u);      // round-to-nearest-even
  unsigned short h = (unsigned short)(r >> 16);
  return __builtin_bit_cast(__bf16, h);
}

// 16x32 bf16 A-tile: lane&15 = M,
// elem e -> K = 16*half + e (contiguous, matches B-tiles)
__device__ __forceinline__ v16bf load_a(const __bf16* row, int half) {
  v8bf lo = *(const v8bf*)(row + 16 * half);
  v8bf hi = *(const v8bf*)(row + 16 * half + 8);
  v16bf r;
#pragma unroll
  for (int i = 0; i < 8; ++i) { r[i] = lo[i]; r[8 + i] = hi[i]; }
  return r;
}

// 16 contiguous bf16 for B-tiles (lane&15 = N, elem e -> K = e + 16*half;
// caller passes pointer already advanced to K = 16*half)
__device__ __forceinline__ v16bf load16(const __bf16* p) {
  v8bf lo = *(const v8bf*)p;
  v8bf hi = *(const v8bf*)(p + 8);
  v16bf r;
#pragma unroll
  for (int i = 0; i < 8; ++i) { r[i] = lo[i]; r[8 + i] = hi[i]; }
  return r;
}

__device__ __forceinline__ v8f wmma_bf16(v16bf a, v16bf b, v8f c) {
  return __builtin_amdgcn_wmma_f32_16x16x32_bf16(
      false, a, false, b, (short)0, c, false, false);
}

// ---------------------------------------------------------------------------
// Elementwise converts
// ---------------------------------------------------------------------------
__global__ void k_f32_to_bf16(const float* __restrict__ src,
                              __bf16* __restrict__ dst, int n) {
  int i = blockIdx.x * blockDim.x + threadIdx.x;
  if (i < n) dst[i] = f2bf(src[i]);
}

// Wt[n][k] = bf16(W[k][n])  (so B-tile K-runs are contiguous)
__global__ void k_transpose_w(const float* __restrict__ W,
                              __bf16* __restrict__ Wt) {
  int i = blockIdx.x * blockDim.x + threadIdx.x;
  if (i < Dd * Dd) {
    int n = i / Dd, k = i % Dd;
    Wt[(size_t)n * Dd + k] = f2bf(W[(size_t)k * Dd + n]);
  }
}

// ---------------------------------------------------------------------------
// Shared GEMM core: one wave, 32x64 output tile, K swept by 64 in a ping-pong
// 2-phase loop: each phase computes one 32-wide K chunk while the loads for
// the next chunk are in flight; register sets alternate, so no copies.
// ---------------------------------------------------------------------------
__device__ __forceinline__ void gemm_core_32x64(
    const __bf16* __restrict__ X, const __bf16* __restrict__ Wt,
    int mbase, int nbase, int lh, int half, v8f acc0[4], v8f acc1[4]) {
  const __bf16* arow0 = X + (size_t)(mbase + lh) * Dd;
  const __bf16* arow1 = X + (size_t)(mbase + 16 + lh) * Dd;
  const __bf16* wrow[4];
#pragma unroll
  for (int nc = 0; nc < 4; ++nc)
    wrow[nc] = Wt + (size_t)(nbase + nc * 16 + lh) * Dd + 16 * half;

  // ping buffers: chunk k0
  v16bf pa0 = load_a(arow0, half);
  v16bf pa1 = load_a(arow1, half);
  v16bf pb[4];
#pragma unroll
  for (int nc = 0; nc < 4; ++nc) pb[nc] = load16(wrow[nc]);

  for (int k0 = 0; k0 < Dd; k0 += 64) {
    // issue loads for chunk k0+32 (always in bounds: k0+32 <= Dd-32)
    v16bf xa0 = load_a(arow0 + k0 + 32, half);
    v16bf xa1 = load_a(arow1 + k0 + 32, half);
    v16bf xb[4];
#pragma unroll
    for (int nc = 0; nc < 4; ++nc) xb[nc] = load16(wrow[nc] + k0 + 32);

    // compute chunk k0 with ping registers
#pragma unroll
    for (int nc = 0; nc < 4; ++nc) {
      acc0[nc] = wmma_bf16(pa0, pb[nc], acc0[nc]);
      acc1[nc] = wmma_bf16(pa1, pb[nc], acc1[nc]);
    }

    // issue loads for chunk k0+64 into ping registers (skip past the end)
    if (k0 + 64 < Dd) {
      pa0 = load_a(arow0 + k0 + 64, half);
      pa1 = load_a(arow1 + k0 + 64, half);
#pragma unroll
      for (int nc = 0; nc < 4; ++nc) pb[nc] = load16(wrow[nc] + k0 + 64);
    }

    // compute chunk k0+32 with pong registers
#pragma unroll
    for (int nc = 0; nc < 4; ++nc) {
      acc0[nc] = wmma_bf16(xa0, xb[nc], acc0[nc]);
      acc1[nc] = wmma_bf16(xa1, xb[nc], acc1[nc]);
    }
  }
}

// Projection GEMM: out_headsplit[b][h][s][dh] = bf16(scale*(X @ W + bias))
__global__ void __launch_bounds__(32)
k_gemm_proj(const __bf16* __restrict__ X, const __bf16* __restrict__ Wt,
            const float* __restrict__ bias, __bf16* __restrict__ outHS,
            float scale) {
  const int lane = threadIdx.x;
  const int lh = lane & 15, half = lane >> 4;
  const int mtile = blockIdx.x / (Dd / 64);
  const int h     = blockIdx.x % (Dd / 64);  // 64-wide n-tile == one head
  const int mbase = mtile * 32;
  const int nbase = h * 64;

  v8f acc0[4] = {}, acc1[4] = {};
  gemm_core_32x64(X, Wt, mbase, nbase, lh, half, acc0, acc1);

#pragma unroll
  for (int nc = 0; nc < 4; ++nc) {
    const int n = nbase + nc * 16 + lh;
    const float bv = bias[n];
    const int dh = nc * 16 + lh;
#pragma unroll
    for (int g = 0; g < 2; ++g) {
#pragma unroll
      for (int r = 0; r < 8; ++r) {
        const int m = mbase + g * 16 + r + 8 * half;
        const int bi = m / Ss, si = m % Ss;
        const float v = ((g ? acc1 : acc0)[nc][r] + bv) * scale;
        outHS[(((size_t)(bi * Hh + h) * Ss + si) * DH) + dh] = f2bf(v);
      }
    }
  }
}

// Final GEMM: d_out[m][n] = ctx @ Wo + bo (fp32 output)
__global__ void __launch_bounds__(32)
k_gemm_out(const __bf16* __restrict__ X, const __bf16* __restrict__ Wt,
           const float* __restrict__ bias, float* __restrict__ out) {
  const int lane = threadIdx.x;
  const int lh = lane & 15, half = lane >> 4;
  const int mtile = blockIdx.x / (Dd / 64);
  const int ntile = blockIdx.x % (Dd / 64);
  const int mbase = mtile * 32;
  const int nbase = ntile * 64;

  v8f acc0[4] = {}, acc1[4] = {};
  gemm_core_32x64(X, Wt, mbase, nbase, lh, half, acc0, acc1);

#pragma unroll
  for (int nc = 0; nc < 4; ++nc) {
    const int n = nbase + nc * 16 + lh;
    const float bv = bias[n];
#pragma unroll
    for (int g = 0; g < 2; ++g) {
#pragma unroll
      for (int r = 0; r < 8; ++r) {
        const int m = mbase + g * 16 + r + 8 * half;
        out[(size_t)m * Dd + n] = (g ? acc1 : acc0)[nc][r] + bv;
      }
    }
  }
}

// ---------------------------------------------------------------------------
// Attention helpers
// ---------------------------------------------------------------------------
__device__ __forceinline__ void load_ktile(const __bf16* __restrict__ Kb,
                                           int kbase, int lh, int half,
                                           v16bf kb[4]) {
  const __bf16* k0r = Kb + (size_t)(kbase + lh) * DH + 16 * half;
  const __bf16* k1r = Kb + (size_t)(kbase + 16 + lh) * DH + 16 * half;
  kb[0] = load16(k0r);        // dh-chunk 0, key-subtile 0
  kb[1] = load16(k1r);        // dh-chunk 0, key-subtile 1
  kb[2] = load16(k0r + 32);   // dh-chunk 1, key-subtile 0
  kb[3] = load16(k1r + 32);   // dh-chunk 1, key-subtile 1
}

// Stage one 32x64 bf16 V tile into an LDS buffer (one key row per lane).
__device__ __forceinline__ void stage_v(const __bf16* __restrict__ Vb,
                                        int kbase, int lane,
                                        __bf16* __restrict__ vbuf) {
  const __bf16* vrow = Vb + (size_t)(kbase + lane) * DH;
#if HAVE_ASYNC_LDS
  // builtin expects v4i pointers: (global AS1 src, LDS AS3 dst, imm off, imm cpol)
  auto gp = (__attribute__((address_space(1))) v4i*)(uintptr_t)(const void*)vrow;
  auto lp = (__attribute__((address_space(3))) v4i*)
                (unsigned)(uintptr_t)(void*)(vbuf + lane * DH);
#pragma unroll
  for (int c = 0; c < 8; ++c)
    __builtin_amdgcn_global_load_async_to_lds_b128(gp + c, lp + c, 0, 0);
#else
#pragma unroll
  for (int c = 0; c < 8; ++c)
    *(v8bf*)&vbuf[lane * DH + c * 8] = *(const v8bf*)(vrow + c * 8);
#endif
}

// ---------------------------------------------------------------------------
// Causal flash attention: one wave per (b, h, 16 query rows).
// Key tiles of 32, pipelined: scores(kt) -> issue loads(kt+1) -> softmax(kt)
// (overlaps loads) -> ctx(kt).
// ---------------------------------------------------------------------------
__global__ void __launch_bounds__(32)
k_attn(const __bf16* __restrict__ Qh, const __bf16* __restrict__ Kh,
       const __bf16* __restrict__ Vh, __bf16* __restrict__ Ctx) {
  __shared__ __align__(16) __bf16 Vlds[2][32 * DH];  // 2 x 4 KB, double buffer
  __shared__ __align__(16) __bf16 Plds[16 * 32];     // 1 KB P re-layout staging

  const int lane = threadIdx.x;
  const int lh = lane & 15, half = lane >> 4;
  const int qtiles = Ss / 16;
  const int bh = blockIdx.x / qtiles;
  const int qbase = (blockIdx.x % qtiles) * 16;
  const int b = bh / Hh, h = bh % Hh;

  const __bf16* Qb = Qh + (size_t)bh * Ss * DH;
  const __bf16* Kb = Kh + (size_t)bh * Ss * DH;
  const __bf16* Vb = Vh + (size_t)bh * Ss * DH;

  // Q A-tiles: two 16x32 K-chunks over DH=64 (Q already scaled by 1/8)
  const __bf16* qrow = Qb + (size_t)(qbase + lh) * DH;
  const v16bf qa0 = load_a(qrow, half);
  const v16bf qa1 = load_a(qrow + 32, half);

  float mrow[8], lrow[8];
#pragma unroll
  for (int r = 0; r < 8; ++r) { mrow[r] = -1e30f; lrow[r] = 0.0f; }
  v8f o[4] = {};

  const int nkt = (qbase + 16 + 31) >> 5;            // causal key-tile bound

  v16bf kcur[4];
  load_ktile(Kb, 0, lh, half, kcur);
  stage_v(Vb, 0, lane, &Vlds[0][0]);

  for (int kt = 0; kt < nkt; ++kt) {
    const int kbase = kt * 32;

    // scores for tile kt (K regs were loaded one iteration ago)
    v8f s0 = {}, s1 = {};
    s0 = wmma_bf16(qa0, kcur[0], s0);
    s1 = wmma_bf16(qa0, kcur[1], s1);
    s0 = wmma_bf16(qa1, kcur[2], s0);
    s1 = wmma_bf16(qa1, kcur[3], s1);

    // issue loads for tile kt+1 (overlap with softmax VALU below);
    // kcur regs were just consumed, so reload in place (no copies)
    const bool more = (kt + 1) < nkt;
    if (more) {
      load_ktile(Kb, kbase + 32, lh, half, kcur);
      stage_v(Vb, kbase + 32, lane, &Vlds[(kt + 1) & 1][0]);
    }
    if (kt + 2 < nkt) {
      __builtin_prefetch(Kb + (size_t)(kbase + 64) * DH, 0, 0);
      __builtin_prefetch(Vb + (size_t)(kbase + 64) * DH, 0, 0);
    }

    // causal mask + online softmax (rows live per lane-half -> xor<16 reduce)
    const int col0 = kbase + lh, col1 = kbase + 16 + lh;
    float p0[8], p1[8], mx[8], rs[8];
#pragma unroll
    for (int r = 0; r < 8; ++r) {
      const int row = qbase + r + 8 * half;
      p0[r] = (col0 <= row) ? s0[r] : -1e30f;
      p1[r] = (col1 <= row) ? s1[r] : -1e30f;
      mx[r] = fmaxf(p0[r], p1[r]);
    }
#pragma unroll
    for (int off = 1; off < 16; off <<= 1)
#pragma unroll
      for (int r = 0; r < 8; ++r)
        mx[r] = fmaxf(mx[r], __shfl_xor(mx[r], off, 32));
#pragma unroll
    for (int r = 0; r < 8; ++r) {
      const float mnew = fmaxf(mrow[r], mx[r]);
      const float alpha = __expf(mrow[r] - mnew);
      mrow[r] = mnew;
      p0[r] = __expf(p0[r] - mnew);
      p1[r] = __expf(p1[r] - mnew);
      rs[r] = p0[r] + p1[r];
      lrow[r] *= alpha;
#pragma unroll
      for (int nc = 0; nc < 4; ++nc) o[nc][r] *= alpha;
    }
#pragma unroll
    for (int off = 1; off < 16; off <<= 1)
#pragma unroll
      for (int r = 0; r < 8; ++r)
        rs[r] += __shfl_xor(rs[r], off, 32);
#pragma unroll
    for (int r = 0; r < 8; ++r) lrow[r] += rs[r];

    // re-layout P (C-layout) -> A-layout via LDS (wave LDS ops are in-order)
#pragma unroll
    for (int r = 0; r < 8; ++r) {
      const int prow = r + 8 * half;
      Plds[prow * 32 + lh]      = f2bf(p0[r]);
      Plds[prow * 32 + 16 + lh] = f2bf(p1[r]);
    }
    const v16bf pa = load_a(&Plds[lh * 32], half);

#if HAVE_ASYNC_LDS
    // ensure tile kt's 8 async chunks landed (kt+1's 8 may stay in flight)
    if (more) ASYNC_WAIT(8); else ASYNC_WAIT(0);
    asm volatile("" ::: "memory");
#endif

    // ctx += P(16x32) . V(32x64): B-tile b[e] = V[e+16*half][nc*16+lh]
    const __bf16* vcur = &Vlds[kt & 1][0];
#pragma unroll
    for (int nc = 0; nc < 4; ++nc) {
      v16bf vb;
#pragma unroll
      for (int e = 0; e < 16; ++e)
        vb[e] = vcur[(e + 16 * half) * DH + nc * 16 + lh];
      o[nc] = wmma_bf16(pa, vb, o[nc]);
    }
  }

  // normalize and store ctx as bf16 in [b][s][h*64+dh] (row-major tokens)
#pragma unroll
  for (int nc = 0; nc < 4; ++nc) {
    const int dfeat = h * DH + nc * 16 + lh;
#pragma unroll
    for (int r = 0; r < 8; ++r) {
      const int m = qbase + r + 8 * half;
      const float v = o[nc][r] / lrow[r];
      Ctx[((size_t)b * Ss + m) * Dd + dfeat] = f2bf(v);
    }
  }
}

// ---------------------------------------------------------------------------
// Host-side orchestration
// ---------------------------------------------------------------------------
extern "C" void kernel_launch(void* const* d_in, const int* in_sizes, int n_in,
                              void* d_out, int out_size, void* d_ws,
                              size_t ws_size, hipStream_t stream) {
  (void)in_sizes; (void)n_in; (void)out_size; (void)ws_size;

  const float* q  = (const float*)d_in[0];
  const float* k  = (const float*)d_in[1];
  const float* v  = (const float*)d_in[2];
  const float* Wq = (const float*)d_in[3];
  const float* bq = (const float*)d_in[4];
  const float* Wk = (const float*)d_in[5];
  const float* bk = (const float*)d_in[6];
  const float* Wv = (const float*)d_in[7];
  const float* bv = (const float*)d_in[8];
  const float* Wo = (const float*)d_in[9];
  const float* bo = (const float*)d_in[10];
  float* out = (float*)d_out;

  char* ws = (char*)d_ws;
  size_t off = 0;
  auto take = [&](size_t bytes) -> char* {
    char* p = ws + off;
    off += (bytes + 255) & ~(size_t)255;
    return p;
  };
  const size_t actB = (size_t)Mm * Dd * sizeof(__bf16);   // 6.29 MB
  const size_t wB   = (size_t)Dd * Dd * sizeof(__bf16);   // 1.18 MB
  __bf16* Xq  = (__bf16*)take(actB);
  __bf16* Xk  = (__bf16*)take(actB);
  __bf16* Xv  = (__bf16*)take(actB);
  __bf16* WqT = (__bf16*)take(wB);
  __bf16* WkT = (__bf16*)take(wB);
  __bf16* WvT = (__bf16*)take(wB);
  __bf16* WoT = (__bf16*)take(wB);
  __bf16* Qh  = (__bf16*)take(actB);
  __bf16* Kh  = (__bf16*)take(actB);
  __bf16* Vh  = (__bf16*)take(actB);
  __bf16* Ctx = (__bf16*)take(actB);

  const int nAct = Mm * Dd;                  // 3,145,728
  const int cvtBlocks = (nAct + 255) / 256;
  k_f32_to_bf16<<<cvtBlocks, 256, 0, stream>>>(q, Xq, nAct);
  k_f32_to_bf16<<<cvtBlocks, 256, 0, stream>>>(k, Xk, nAct);
  k_f32_to_bf16<<<cvtBlocks, 256, 0, stream>>>(v, Xv, nAct);

  const int wBlocks = (Dd * Dd + 255) / 256;
  k_transpose_w<<<wBlocks, 256, 0, stream>>>(Wq, WqT);
  k_transpose_w<<<wBlocks, 256, 0, stream>>>(Wk, WkT);
  k_transpose_w<<<wBlocks, 256, 0, stream>>>(Wv, WvT);
  k_transpose_w<<<wBlocks, 256, 0, stream>>>(Wo, WoT);

  const int gemmBlocks = (Mm / 32) * (Dd / 64);   // 1536
  const float qscale = 0.125f;                    // 1/sqrt(DH)
  k_gemm_proj<<<gemmBlocks, 32, 0, stream>>>(Xq, WqT, bq, Qh, qscale);
  k_gemm_proj<<<gemmBlocks, 32, 0, stream>>>(Xk, WkT, bk, Kh, 1.0f);
  k_gemm_proj<<<gemmBlocks, 32, 0, stream>>>(Xv, WvT, bv, Vh, 1.0f);

  const int attnBlocks = Bb * Hh * (Ss / 16);     // 3072
  k_attn<<<attnBlocks, 32, 0, stream>>>(Qh, Kh, Vh, Ctx);

  k_gemm_out<<<gemmBlocks, 32, 0, stream>>>(Ctx, WoT, bo, out);
}
